// LocalGATEncoder_74191265071675
// MI455X (gfx1250) — matrix operand order repaired and checked by reading, not compiled, not run on hardware
//
#include <hip/hip_runtime.h>
#include <math.h>

// ---------------------------------------------------------------------------
// LocalGATEncoder for MI455X (gfx1250): 3-layer GAT, D=128, H=8.
// GEMM h@Wg runs on v_wmma_f32_16x16x32_f16 (f16 in, f32 accumulate) with
// explicit double-buffered B fragments so B loads overlap WMMA issue;
// edge softmax/aggregation uses float atomics; LN uses wave32 shuffles.
// ---------------------------------------------------------------------------

typedef __attribute__((ext_vector_type(16))) _Float16 v16h;
typedef __attribute__((ext_vector_type(8)))  _Float16 v8h;
typedef __attribute__((ext_vector_type(8)))  float    v8f;

#define N_NODES 20000
#define N_EDGES 200000
#define DIN     10
#define DM      128
#define NH      8
#define NL      3
#define HD      1024   // NH * DM
#define NEG_SLOPE 0.2f
#define LN_EPS    1e-5f

// ---------------- helpers ----------------
__device__ __forceinline__ float warp_sum32(float v) {
#pragma unroll
    for (int off = 16; off > 0; off >>= 1) v += __shfl_xor(v, off, 32);
    return v;
}

__device__ __forceinline__ float gelu_exact(float x) {
    return 0.5f * x * (1.0f + erff(x * 0.70710678118654752f));
}

__device__ __forceinline__ void atomicMaxF(float* addr, float val) {
    if (val >= 0.0f) atomicMax((int*)addr, __float_as_int(val));
    else             atomicMin((unsigned int*)addr, __float_as_uint(val));
}

// ---------------- Wg [L,128,1024] f32 -> WgT [L,1024,128] f16 ----------------
__global__ void k_wg_transpose(const float* __restrict__ Wg, _Float16* __restrict__ wgT,
                               int total) {
    int i = blockIdx.x * blockDim.x + threadIdx.x;
    if (i >= total) return;                 // total = NL*DM*HD
    int l   = i / (DM * HD);
    int rem = i - l * (DM * HD);
    int k   = rem / HD;                     // 0..127
    int n   = rem - k * HD;                 // 0..1023
    wgT[(size_t)l * HD * DM + (size_t)n * DM + k] = (_Float16)Wg[i];
}

// ---------------- input projection + LN + GELU (wave per node) ----------------
__global__ void k_input_proj(const float* __restrict__ x, const float* __restrict__ Win,
                             const float* __restrict__ bin, const float* __restrict__ g,
                             const float* __restrict__ b, float* __restrict__ h,
                             _Float16* __restrict__ h16, int n_nodes) {
    int node = (blockIdx.x * blockDim.x + threadIdx.x) >> 5;
    int lane = threadIdx.x & 31;
    if (node >= n_nodes) return;
    float xr[DIN];
#pragma unroll
    for (int k = 0; k < DIN; ++k) xr[k] = x[node * DIN + k];
    float v[4];
#pragma unroll
    for (int j = 0; j < 4; ++j) {
        int d = lane + j * 32;
        float s = bin[d];
#pragma unroll
        for (int k = 0; k < DIN; ++k) s += xr[k] * Win[k * DM + d];
        v[j] = s;
    }
    float mu = warp_sum32(v[0] + v[1] + v[2] + v[3]) * (1.0f / DM);
    float var = 0.0f;
#pragma unroll
    for (int j = 0; j < 4; ++j) { float t = v[j] - mu; var += t * t; }
    var = warp_sum32(var) * (1.0f / DM);
    float rs = rsqrtf(var + LN_EPS);
#pragma unroll
    for (int j = 0; j < 4; ++j) {
        int d = lane + j * 32;
        float y = gelu_exact((v[j] - mu) * rs * g[d] + b[d]);
        h[(size_t)node * DM + d]   = y;
        h16[(size_t)node * DM + d] = (_Float16)y;
    }
}

// ---------------- hW = h16[N,128] @ WgT(B)[1024,128]^T -> [N,1024] ----------------
// Block: 256 threads = 8 waves; block owns a 16-row strip; each wave owns 128 cols.
// B fragments are double-buffered so tile c+1's loads fly during tile c's WMMAs.
__global__ void __launch_bounds__(256)
k_gemm_wmma(const _Float16* __restrict__ A16, const _Float16* __restrict__ BT16,
            float* __restrict__ C) {
    const int wave = threadIdx.x >> 5;
    const int lane = threadIdx.x & 31;
    const int row0 = blockIdx.x * 16;

    // ---- A fragments: contiguous-K 16x32 layout (k = 16*(lane/16) + s, same order as B) ----
    // lanes 0-15 : row M=lane,    K = {0..15}
    // lanes 16-31: row M=lane-16, K = {16..31}
    const int rA = lane & 15;
    const int kA = (lane & 16) ? 16 : 0;
    v16h a[4];
#pragma unroll
    for (int kk = 0; kk < 4; ++kk) {
        const _Float16* p = A16 + (size_t)(row0 + rA) * DM + kk * 32 + kA;
        v8h lo = *(const v8h*)p;
        v8h hi = *(const v8h*)(p + 8);
#pragma unroll
        for (int i = 0; i < 8; ++i) { a[kk][i] = lo[i]; a[kk][8 + i] = hi[i]; }
    }

    // ---- B fragments: lanes 0-15 hold K 0..15 of column N=lane, lanes 16-31 K 16..31
    const int nB = lane & 15;
    const int kB = (lane & 16) ? 16 : 0;
    const _Float16* qBase = BT16 + (size_t)(wave * 128 + nB) * DM + kB;

    const int nOut  = lane & 15;
    const int mBase = (lane >> 4) * 8;

    v16h bf[2][4];
#pragma unroll
    for (int kk = 0; kk < 4; ++kk)          // preload tile c = 0
        bf[0][kk] = *(const v16h*)(qBase + kk * 32);

#pragma unroll
    for (int c = 0; c < 8; ++c) {
        const int cur = c & 1;
        if (c < 7) {                        // issue next tile's B loads first
            const _Float16* qn = qBase + (size_t)(c + 1) * 16 * DM;
            __builtin_prefetch(qn + 16 * DM, 0, 1);   // -> global_prefetch_b8
#pragma unroll
            for (int kk = 0; kk < 4; ++kk)
                bf[cur ^ 1][kk] = *(const v16h*)(qn + kk * 32);
        }
        v8f acc = {};
#pragma unroll
        for (int kk = 0; kk < 4; ++kk) {
            acc = __builtin_amdgcn_wmma_f32_16x16x32_f16(
                /*neg_a=*/false, a[kk], /*neg_b=*/false, bf[cur][kk],
                /*c_mod=*/(short)0, acc, /*reuse_a=*/false, /*reuse_b=*/false);
        }
        const size_t cbase = (size_t)(row0 + mBase) * HD
                           + (size_t)(wave * 128 + c * 16) + nOut;
#pragma unroll
        for (int r = 0; r < 8; ++r)
            C[cbase + (size_t)r * HD] = acc[r];
    }
}

// ---------------- per-node attention logits es/ed (thread per (n,h)) ----------------
__global__ void k_scores(const float* __restrict__ hW, const float* __restrict__ a_s,
                         const float* __restrict__ a_d, float* __restrict__ es,
                         float* __restrict__ ed, int n_nodes) {
    int idx = blockIdx.x * blockDim.x + threadIdx.x;
    if (idx >= n_nodes * NH) return;
    int n  = idx >> 3;
    int hh = idx & 7;
    const float4* p   = (const float4*)(hW + (size_t)n * HD + hh * DM);
    const float4* as4 = (const float4*)(a_s + hh * DM);
    const float4* ad4 = (const float4*)(a_d + hh * DM);
    float s1 = 0.0f, s2 = 0.0f;
#pragma unroll 8
    for (int k = 0; k < DM / 4; ++k) {
        float4 hv = p[k], av = as4[k], dv = ad4[k];
        s1 += hv.x * av.x + hv.y * av.y + hv.z * av.z + hv.w * av.w;
        s2 += hv.x * dv.x + hv.y * dv.y + hv.z * dv.z + hv.w * dv.w;
    }
    es[idx] = s1;
    ed[idx] = s2;
}

// ---------------- zero/neg-inf initialization for the segment ops ----------------
__global__ void k_init(float* __restrict__ acc, float* __restrict__ m,
                       float* __restrict__ den, int n_nodes) {
    int i = blockIdx.x * blockDim.x + threadIdx.x;
    if (i < n_nodes * DM) acc[i] = 0.0f;
    if (i < n_nodes * NH) { m[i] = -__builtin_inff(); den[i] = 0.0f; }
}

// ---------------- edge logits + segment max (thread per edge) ----------------
__global__ void k_edge_logits(const int* __restrict__ src, const int* __restrict__ dst,
                              const float* __restrict__ es, const float* __restrict__ ed,
                              float* __restrict__ est, float* __restrict__ m, int n_edges) {
    int e = blockIdx.x * blockDim.x + threadIdx.x;
    if (e >= n_edges) return;
    int s = src[e], d = dst[e];
#pragma unroll
    for (int hh = 0; hh < NH; ++hh) {
        float v = es[s * NH + hh] + ed[d * NH + hh];
        v = (v >= 0.0f) ? v : NEG_SLOPE * v;
        est[(size_t)e * NH + hh] = v;
        atomicMaxF(&m[d * NH + hh], v);
    }
}

// ---------------- exp(e - m) + segment sum of denominators ----------------
__global__ void k_edge_exp(const int* __restrict__ dst, float* __restrict__ est,
                           const float* __restrict__ m, float* __restrict__ den,
                           int n_edges) {
    int e = blockIdx.x * blockDim.x + threadIdx.x;
    if (e >= n_edges) return;
    int d = dst[e];
#pragma unroll
    for (int hh = 0; hh < NH; ++hh) {
        float num = expf(est[(size_t)e * NH + hh] - m[d * NH + hh]);
        est[(size_t)e * NH + hh] = num;
        atomicAdd(&den[d * NH + hh], num);
    }
}

// ---------------- alpha * hW[src] segment-sum (block of 128 per edge) ----------------
__global__ void __launch_bounds__(128)
k_edge_accum(const int* __restrict__ src, const int* __restrict__ dst,
             const float* __restrict__ est, const float* __restrict__ den,
             const float* __restrict__ hW, float* __restrict__ acc,
             float* __restrict__ alpha_out) {
    int e = blockIdx.x;
    int d = threadIdx.x;
    int sn = src[e], dn = dst[e];
    float alpha[NH];
    float asum = 0.0f;
#pragma unroll
    for (int hh = 0; hh < NH; ++hh) {
        alpha[hh] = est[(size_t)e * NH + hh] / (den[dn * NH + hh] + 1e-16f);
        asum += alpha[hh];
    }
    float s = 0.0f;
#pragma unroll
    for (int hh = 0; hh < NH; ++hh)
        s += alpha[hh] * hW[(size_t)sn * HD + hh * DM + d];
    atomicAdd(&acc[(size_t)dn * DM + d], s * 0.125f);          // head mean
    if (d == 0 && alpha_out) alpha_out[e] = asum * 0.125f;     // alpha.mean(axis=1)
}

// ---------------- residual + LN (+GELU) per node (wave per node) ----------------
__global__ void k_ln_post(const float* __restrict__ acc, const float* __restrict__ bg,
                          const float* __restrict__ hin, const float* __restrict__ g,
                          const float* __restrict__ b, float* __restrict__ hout,
                          _Float16* __restrict__ h16, int n_nodes, int apply_gelu) {
    int node = (blockIdx.x * blockDim.x + threadIdx.x) >> 5;
    int lane = threadIdx.x & 31;
    if (node >= n_nodes) return;
    float v[4];
#pragma unroll
    for (int j = 0; j < 4; ++j) {
        int d = lane + j * 32;
        v[j] = acc[(size_t)node * DM + d] + bg[d] + hin[(size_t)node * DM + d];
    }
    float mu = warp_sum32(v[0] + v[1] + v[2] + v[3]) * (1.0f / DM);
    float var = 0.0f;
#pragma unroll
    for (int j = 0; j < 4; ++j) { float t = v[j] - mu; var += t * t; }
    var = warp_sum32(var) * (1.0f / DM);
    float rs = rsqrtf(var + LN_EPS);
#pragma unroll
    for (int j = 0; j < 4; ++j) {
        int d = lane + j * 32;
        float y = (v[j] - mu) * rs * g[d] + b[d];
        if (apply_gelu) y = gelu_exact(y);
        hout[(size_t)node * DM + d] = y;
        if (h16) h16[(size_t)node * DM + d] = (_Float16)y;
    }
}

// ---------------------------------------------------------------------------
extern "C" void kernel_launch(void* const* d_in, const int* in_sizes, int n_in,
                              void* d_out, int out_size, void* d_ws, size_t ws_size,
                              hipStream_t stream) {
    (void)in_sizes; (void)n_in; (void)out_size; (void)ws_size;

    const float* x    = (const float*)d_in[0];
    const int*   ei   = (const int*)d_in[1];     // [2,E] int32
    const float* Win  = (const float*)d_in[2];
    const float* bin  = (const float*)d_in[3];
    const float* gin  = (const float*)d_in[4];
    const float* bein = (const float*)d_in[5];
    const float* Wg   = (const float*)d_in[6];   // [L,128,1024]
    const float* a_s  = (const float*)d_in[7];   // [L,8,128]
    const float* a_d  = (const float*)d_in[8];
    const float* b_g  = (const float*)d_in[9];   // [L,128]
    const float* lng  = (const float*)d_in[10];
    const float* lnb  = (const float*)d_in[11];

    const int* src = ei;
    const int* dst = ei + N_EDGES;

    float* out_h     = (float*)d_out;                       // [N,128]
    float* out_alpha = out_h + (size_t)N_NODES * DM;        // [E]

    // workspace carve-out
    float* ws   = (float*)d_ws;
    float* h    = ws;  ws += (size_t)N_NODES * DM;
    float* hW   = ws;  ws += (size_t)N_NODES * HD;
    float* es   = ws;  ws += (size_t)N_NODES * NH;
    float* ed   = ws;  ws += (size_t)N_NODES * NH;
    float* mbuf = ws;  ws += (size_t)N_NODES * NH;
    float* den  = ws;  ws += (size_t)N_NODES * NH;
    float* accb = ws;  ws += (size_t)N_NODES * DM;
    float* est  = ws;  ws += (size_t)N_EDGES * NH;
    _Float16* h16 = (_Float16*)ws;
    _Float16* wgT = h16 + (size_t)N_NODES * DM;             // [L,1024,128] f16

    // Wg -> transposed f16 (one-time per launch; L2-resident afterwards)
    {
        int total = NL * DM * HD;
        k_wg_transpose<<<(total + 255) / 256, 256, 0, stream>>>(Wg, wgT, total);
    }

    // input projection + LN + GELU
    k_input_proj<<<(N_NODES + 7) / 8, 256, 0, stream>>>(x, Win, bin, gin, bein,
                                                        h, h16, N_NODES);

    for (int l = 0; l < NL; ++l) {
        const bool last = (l == NL - 1);

        // hW = h @ Wg[l]  (WMMA, f16 in / f32 acc)
        k_gemm_wmma<<<N_NODES / 16, 256, 0, stream>>>(
            h16, wgT + (size_t)l * HD * DM, hW);

        // per-node per-head logits
        k_scores<<<(N_NODES * NH + 255) / 256, 256, 0, stream>>>(
            hW, a_s + (size_t)l * NH * DM, a_d + (size_t)l * NH * DM, es, ed, N_NODES);

        // reset segment buffers
        k_init<<<((size_t)N_NODES * DM + 255) / 256, 256, 0, stream>>>(
            accb, mbuf, den, N_NODES);

        // edge softmax (max, exp, sum) + aggregation
        k_edge_logits<<<(N_EDGES + 255) / 256, 256, 0, stream>>>(
            src, dst, es, ed, est, mbuf, N_EDGES);
        k_edge_exp<<<(N_EDGES + 255) / 256, 256, 0, stream>>>(
            dst, est, mbuf, den, N_EDGES);
        k_edge_accum<<<N_EDGES, 128, 0, stream>>>(
            src, dst, est, den, hW, accb, last ? out_alpha : nullptr);

        // residual + LN (+GELU except last); last layer writes straight to d_out
        k_ln_post<<<(N_NODES + 7) / 8, 256, 0, stream>>>(
            accb, b_g + (size_t)l * DM, h, lng + (size_t)l * DM, lnb + (size_t)l * DM,
            last ? out_h : h, last ? nullptr : h16, N_NODES, last ? 0 : 1);
    }
}
